// BailingMoELinearAttention_73667279061187
// MI455X (gfx1250) — matrix operand order repaired and checked by reading, not compiled, not run on hardware
//
#include <hip/hip_runtime.h>
#include <hip/hip_bf16.h>
#include <math.h>

// ---------------- problem constants (from reference) ----------------
#define HIDDIM 2048
#define NH     16
#define HD     128
#define CHUNK  256
#define NTOK   8192
#define NCHUNK 32            // NTOK / CHUNK
#define QKVW   (3 * NH * HD) // 6144

typedef __bf16 bf16_t;
typedef __attribute__((ext_vector_type(16))) __bf16 v16bf;
typedef __attribute__((ext_vector_type(8)))  __bf16 v8bf;
typedef __attribute__((ext_vector_type(8)))  float  v8f;

__device__ __forceinline__ int lane_id() { return (int)(threadIdx.x & 31); }

__device__ __forceinline__ v8f vzero() {
  v8f z = {0.f, 0.f, 0.f, 0.f, 0.f, 0.f, 0.f, 0.f};
  return z;
}

// slopes: start = 2^-0.5 for H=16; s_h = start^(h+1) * (1 - 0/19 + 1e-5)
__device__ __forceinline__ float head_slope(int h) {
  return exp2f(-0.5f * (float)(h + 1)) * 1.00001f;
}

__device__ __forceinline__ v8bf cvt8(float4 a, float4 b) {
  v8bf r;
  r[0] = (bf16_t)a.x; r[1] = (bf16_t)a.y; r[2] = (bf16_t)a.z; r[3] = (bf16_t)a.w;
  r[4] = (bf16_t)b.x; r[5] = (bf16_t)b.y; r[6] = (bf16_t)b.z; r[7] = (bf16_t)b.w;
  return r;
}
__device__ __forceinline__ v8bf cvt8a(const float* t) {
  v8bf r;
#pragma unroll
  for (int i = 0; i < 8; ++i) r[i] = (bf16_t)t[i];
  return r;
}

__device__ __forceinline__ v8f wmma_bf16(v16bf a, v16bf b, v8f c) {
  return __builtin_amdgcn_wmma_f32_16x16x32_bf16(false, a, false, b, (short)0, c,
                                                 false, false);
}

// Fragment-major LDS layout (ISA 7.12.2, wave32):
//   each 16x32 A tile / 32x16 B tile is stored as [lane 0..31][16 bf16],
//   i.e. 32 contiguous bytes per lane -> one v16bf load (2x ds_load_b128).
// A-layout element (m, k):  lane = (m&15) + 16*((k>>3)&1),
//                           pos  = 8*(k>>4) + (k&7)
//   -> a global-contiguous K-octet j (k = 8j..8j+7) of row m is one contiguous
//      16-byte chunk: lane = (m&15)+16*(j&1), byte-half = j>>1.
// B-layout element (k, n):  lane = (n&15) + 16*((k>>3)&1),
//                           pos  = 8*(k>>4) + (k&7)
//   -> K-octet j (k = 8j..8j+7) of column n is one contiguous 16-byte chunk:
//      lane = (n&15)+16*(j&1), byte-half = j>>1.
__device__ __forceinline__ v16bf frag_ld(const bf16_t* fbase, int tile) {
  return *(const v16bf*)(fbase + (size_t)(tile * 32 + lane_id()) * 16);
}
__device__ __forceinline__ void frag_st8(bf16_t* fbase, int tile, int lane,
                                         int half, v8bf v) {
  *(v8bf*)(fbase + (size_t)(tile * 32 + lane) * 16 + 8 * half) = v;
}

// ---------------- generic f32->bf16 GEMM: C[M,N] = A[M,K] @ B[K,N] --------------
// N, K compile-time so strided loads use immediate offsets (no per-load v_add).
// grid = (N/128, M/128), block = 256 threads (8 waves, each 32x64 output tile)
template <int N, int K>
__global__ __launch_bounds__(256) void gemm_bf16_kernel(
    const float* __restrict__ A, const float* __restrict__ B,
    float* __restrict__ C) {
  __shared__ __align__(32) bf16_t Af[8 * 32 * 16];  // 8 m-tiles, frag-major, 8KB
  __shared__ __align__(32) bf16_t Bf[8 * 32 * 16];  // 8 n-tiles, frag-major, 8KB
  const int tid = threadIdx.x;
  const int wid = tid >> 5;
  const int mt0 = 2 * (wid & 3);        // wave's two m-tiles
  const int nt0 = 4 * (wid >> 2);       // wave's four n-tiles
  const int m0  = blockIdx.y * 128;
  const int n0  = blockIdx.x * 128;

  v8f acc[2][4];
#pragma unroll
  for (int i = 0; i < 2; ++i)
#pragma unroll
    for (int j = 0; j < 4; ++j) acc[i][j] = vzero();

  for (int k0 = 0; k0 < K; k0 += 32) {
    // stage A 128x32: thread -> (row r, K-octet j); contiguous 32B global read,
    // one 16B fragment-major LDS store.
#pragma unroll
    for (int it = 0; it < 2; ++it) {
      const int idx = tid + it * 256;          // 512 octets
      const int j = idx & 3, r = idx >> 2;
      const float* ap = &A[(size_t)(m0 + r) * K + (k0 + 8 * j)];
      const float4 x = *(const float4*)ap;
      const float4 y = *(const float4*)(ap + 4);
      frag_st8(Af, r >> 4, (r & 15) + 16 * (j & 1), j >> 1, cvt8(x, y));
    }
    // stage B 32x128: thread -> (col, K-octet jj); 8 lane-coalesced strided
    // reads (immediate offsets u*N*4), one 16B fragment-major LDS store.
#pragma unroll
    for (int it = 0; it < 2; ++it) {
      const int idx = tid + it * 256;
      const int col = idx & 127, jj = idx >> 7;
      const float* bp = &B[(size_t)(k0 + 8 * jj) * N + (n0 + col)];
      float t[8];
#pragma unroll
      for (int u = 0; u < 8; ++u) t[u] = bp[(size_t)u * N];
      frag_st8(Bf, col >> 4, (col & 15) + 16 * (jj & 1), jj >> 1, cvt8a(t));
    }
    if (k0 + 32 < K) {  // hint next K-tile into cache (global_prefetch_b8)
      __builtin_prefetch(&A[(size_t)(m0 + (tid >> 1)) * K + (k0 + 32)], 0, 3);
      __builtin_prefetch(&B[(size_t)(k0 + 32 + (tid >> 3)) * N + n0], 0, 3);
    }
    __syncthreads();
    const v16bf a0 = frag_ld(Af, mt0);
    const v16bf a1 = frag_ld(Af, mt0 + 1);
#pragma unroll
    for (int j = 0; j < 4; ++j) {
      const v16bf b = frag_ld(Bf, nt0 + j);
      acc[0][j] = wmma_bf16(a0, b, acc[0][j]);
      acc[1][j] = wmma_bf16(a1, b, acc[1][j]);
    }
    __syncthreads();
  }
  // C/D layout: row = (lane>>4)*8 + r, col = lane&15
  const int lane = lane_id();
  const int cr = (lane >> 4) * 8;
  const int cc = lane & 15;
#pragma unroll
  for (int i = 0; i < 2; ++i)
#pragma unroll
    for (int j = 0; j < 4; ++j)
#pragma unroll
      for (int r = 0; r < 8; ++r) {
        const int row = m0 + (mt0 + i) * 16 + cr + r;
        const int col = n0 + (nt0 + j) * 16 + cc;
        C[(size_t)row * N + col] = acc[i][j][r];
      }
}

// ---------------- RoPE on q,k in qkv scratch; q *= D^-0.5 ----------------
__global__ __launch_bounds__(256) void rope_kernel(float* __restrict__ qkv,
                                                   const int* __restrict__ pos) {
  const int idx = blockIdx.x * 256 + threadIdx.x;  // N*H*64
  const int d2 = idx & 63;
  const int h  = (idx >> 6) & 15;
  const int n  = idx >> 10;
  if (n >= NTOK) return;
  const float inv = __powf(600000.0f, -(float)d2 * (1.0f / 64.0f));
  const float p = (float)pos[n];
  float sn, cs;
  __sincosf(p * inv, &sn, &cs);
  float* q = qkv + (size_t)n * QKVW + h * HD;
  float* k = q + NH * HD;
  const float qs = 0.08838834764831845f;  // 128^-0.5
  const float q1 = q[d2], q2 = q[d2 + 64];
  q[d2]      = (q1 * cs - q2 * sn) * qs;
  q[d2 + 64] = (q2 * cs + q1 * sn) * qs;
  const float k1 = k[d2], k2 = k[d2 + 64];
  k[d2]      = k1 * cs - k2 * sn;
  k[d2 + 64] = k2 * cs + k1 * sn;
}

// ---------------- phase 1: kvc[c,h] = (k * k_decay)^T @ v  (128x128) ----------
// grid = (NH, NCHUNK), block = 256
__global__ __launch_bounds__(256) void kv_chunk_kernel(
    const float* __restrict__ qkv, float* __restrict__ kvc) {
  const int h = blockIdx.x;
  const int c = blockIdx.y;
  const float s = head_slope(h);
  __shared__ __align__(32) bf16_t Af[8 * 32 * 16];  // A[d][nn] = k[n][d]*dec
  __shared__ __align__(32) bf16_t Bf[8 * 32 * 16];  // B[nn][e] = v[n][e]
  const int tid = threadIdx.x;
  const int wid = tid >> 5;
  const int mt0 = 2 * (wid & 3);
  const int nt0 = 4 * (wid >> 2);

  v8f acc[2][4];
#pragma unroll
  for (int i = 0; i < 2; ++i)
#pragma unroll
    for (int j = 0; j < 4; ++j) acc[i][j] = vzero();

  const float* kb = qkv + (size_t)(c * CHUNK) * QKVW + NH * HD + h * HD;
  const float* vb = qkv + (size_t)(c * CHUNK) * QKVW + 2 * NH * HD + h * HD;

  for (int n0 = 0; n0 < CHUNK; n0 += 32) {
    // A = k^T (m = d, K = n): strided lane-coalesced reads (imm offsets),
    // per-n decay, one 16B fragment store.
#pragma unroll
    for (int it = 0; it < 2; ++it) {
      const int idx = tid + it * 256;
      const int d = idx & 127, j = idx >> 7;
      const float* kp = &kb[(size_t)(n0 + 8 * j) * QKVW + d];
      float t[8];
#pragma unroll
      for (int u = 0; u < 8; ++u) {
        const int n = n0 + 8 * j + u;
        t[u] = kp[(size_t)u * QKVW] * __expf(-s * (float)(CHUNK - 1 - n));
      }
      frag_st8(Af, d >> 4, (d & 15) + 16 * (j & 1), j >> 1, cvt8a(t));
    }
    // B = v (K = n, col = e)
#pragma unroll
    for (int it = 0; it < 2; ++it) {
      const int idx = tid + it * 256;
      const int e = idx & 127, jj = idx >> 7;
      const float* vp = &vb[(size_t)(n0 + 8 * jj) * QKVW + e];
      float t[8];
#pragma unroll
      for (int u = 0; u < 8; ++u) t[u] = vp[(size_t)u * QKVW];
      frag_st8(Bf, e >> 4, (e & 15) + 16 * (jj & 1), jj >> 1, cvt8a(t));
    }
    __syncthreads();
    const v16bf a0 = frag_ld(Af, mt0);
    const v16bf a1 = frag_ld(Af, mt0 + 1);
#pragma unroll
    for (int j = 0; j < 4; ++j) {
      const v16bf b = frag_ld(Bf, nt0 + j);
      acc[0][j] = wmma_bf16(a0, b, acc[0][j]);
      acc[1][j] = wmma_bf16(a1, b, acc[1][j]);
    }
    __syncthreads();
  }
  float* o = kvc + ((size_t)c * NH + h) * (HD * HD);
  const int lane = lane_id();
  const int cr = (lane >> 4) * 8;
  const int cc = lane & 15;
#pragma unroll
  for (int i = 0; i < 2; ++i)
#pragma unroll
    for (int j = 0; j < 4; ++j)
#pragma unroll
      for (int r = 0; r < 8; ++r) {
        const int d = (mt0 + i) * 16 + cr + r;
        const int e = (nt0 + j) * 16 + cc;
        o[(size_t)d * HD + e] = acc[i][j][r];
      }
}

// ---------------- phase 2: kv_in[c] = bd*kv_in[c-1] + kvc[c-1], kv_in[0]=0 ------
__global__ __launch_bounds__(256) void kv_scan_kernel(
    const float* __restrict__ kvc, float* __restrict__ kvin) {
  const int idx = blockIdx.x * 256 + threadIdx.x;  // NH * 128*128
  const int h = idx >> 14;
  const int j = idx & 16383;
  const float s  = head_slope(h);
  const float bd = __expf(-s * (float)CHUNK);
  float acc = 0.0f;
  for (int c = 0; c < NCHUNK; ++c) {
    const size_t off = ((size_t)c * NH + h) * (HD * HD) + j;
    kvin[off] = acc;             // state *before* chunk c
    acc = bd * acc + kvc[off];
  }
}

// ---------------- phase 3: O = rowdiag(q_decay)*(q@kv_in) + (mask∘(q@k^T))@v ----
// grid = (NH, NCHUNK), block = 256; full chunk resident in 240KB LDS,
// all operands pre-arranged fragment-major.
__global__ __launch_bounds__(256) void attn_out_kernel(
    const float* __restrict__ qkv, const float* __restrict__ kvin,
    float* __restrict__ attn) {
  extern __shared__ __align__(32) bf16_t sm[];
  bf16_t* qA  = sm;                        // A-frags: 16 m-tiles x 4 k-steps (64KB)
  bf16_t* kB  = qA + 16 * 4 * 32 * 16;     // B^T-frags: 4 k-steps x 16 n-tiles (64KB)
  bf16_t* vB  = kB + 4 * 16 * 32 * 16;     // B-frags: 8 k-steps x 8 n-tiles (64KB)
  bf16_t* kvB = vB + 8 * 8 * 32 * 16;      // B-frags: 4 k-steps x 8 n-tiles (32KB)
  bf16_t* swA = kvB + 4 * 8 * 32 * 16;     // per-wave scores: 8 x 2 tiles (16KB)

  const int h = blockIdx.x;
  const int c = blockIdx.y;
  const float s = head_slope(h);
  const int tid = threadIdx.x;

  const float* qb = qkv + (size_t)(c * CHUNK) * QKVW + h * HD;
  const float* kb = qb + NH * HD;
  const float* vb = qb + 2 * NH * HD;

  // q -> A-frag layout: thread (token n, d-octet j); contiguous 32B reads.
#pragma unroll
  for (int it = 0; it < 16; ++it) {
    const int idx = tid + it * 256;          // 4096
    const int j = idx & 15, n = idx >> 4;
    const float* p = &qb[(size_t)n * QKVW + 8 * j];
    const v8bf v = cvt8(*(const float4*)p, *(const float4*)(p + 4));
    const int kstep = j >> 2, jo = j & 3;
    frag_st8(qA, (n >> 4) * 4 + kstep, (n & 15) + 16 * (jo & 1), jo >> 1, v);
  }
  // k -> B^T-frag layout (logical B[d][n] = k[n][d]): contiguous 32B reads.
#pragma unroll
  for (int it = 0; it < 16; ++it) {
    const int idx = tid + it * 256;
    const int j = idx & 15, n = idx >> 4;
    const float* p = &kb[(size_t)n * QKVW + 8 * j];
    const v8bf v = cvt8(*(const float4*)p, *(const float4*)(p + 4));
    const int kstep = j >> 2, jo = j & 3;
    frag_st8(kB, kstep * 16 + (n >> 4), (n & 15) + 16 * (jo & 1), jo >> 1, v);
  }
  // v -> B-frag layout (K = token n, col = e): lane-coalesced strided reads.
#pragma unroll
  for (int it = 0; it < 16; ++it) {
    const int idx = tid + it * 256;
    const int e = idx & 127, nj = idx >> 7;  // nj: n-octet 0..31
    const float* vp = &vb[(size_t)(8 * nj) * QKVW + e];
    float t[8];
#pragma unroll
    for (int u = 0; u < 8; ++u) t[u] = vp[(size_t)u * QKVW];
    const int kstep = nj >> 2, jo = nj & 3;
    frag_st8(vB, kstep * 8 + (e >> 4), (e & 15) + 16 * (jo & 1), jo >> 1, cvt8a(t));
  }
  // kv_in -> B-frag layout (K = d, col = e)
  const float* kvb = kvin + ((size_t)c * NH + h) * (HD * HD);
#pragma unroll
  for (int it = 0; it < 8; ++it) {
    const int idx = tid + it * 256;          // 2048
    const int e = idx & 127, dj = idx >> 7;  // dj: d-octet 0..15
    const float* kp = &kvb[(size_t)(8 * dj) * HD + e];
    float t[8];
#pragma unroll
    for (int u = 0; u < 8; ++u) t[u] = kp[(size_t)u * HD];
    const int kstep = dj >> 2, jo = dj & 3;
    frag_st8(kvB, kstep * 8 + (e >> 4), (e & 15) + 16 * (jo & 1), jo >> 1, cvt8a(t));
  }
  __syncthreads();

  const int wid  = tid >> 5;
  const int m0   = wid * 32;               // wave owns rows [m0, m0+32)
  const int lane = lane_id();
  const int cr = (lane >> 4) * 8;
  const int cc = lane & 15;

  v8f o[2][8];
#pragma unroll
  for (int i = 0; i < 2; ++i)
#pragma unroll
    for (int j = 0; j < 8; ++j) o[i][j] = vzero();

  // inter-chunk: O = q @ kv_in  (K = 128)
#pragma unroll
  for (int kk = 0; kk < 4; ++kk) {
    const v16bf a0 = frag_ld(qA, (2 * wid) * 4 + kk);
    const v16bf a1 = frag_ld(qA, (2 * wid + 1) * 4 + kk);
#pragma unroll
    for (int j = 0; j < 8; ++j) {
      const v16bf b = frag_ld(kvB, kk * 8 + j);
      o[0][j] = wmma_bf16(a0, b, o[0][j]);
      o[1][j] = wmma_bf16(a1, b, o[1][j]);
    }
  }
  // row-diagonal q_decay commutes: scale accumulators by exp(-s*(m+1))
#pragma unroll
  for (int i = 0; i < 2; ++i)
#pragma unroll
    for (int r = 0; r < 8; ++r) {
      const int m = m0 + i * 16 + cr + r;
      const float qd = __expf(-s * (float)(m + 1));
#pragma unroll
      for (int j = 0; j < 8; ++j) o[i][j][r] *= qd;
    }

  // intra-chunk: causal key tiles t <= wid.
  // Decay factorization within a tile: exp(-s*(m-n)) = rowf(m,t) * colf(n-32t),
  // rowf = exp(-s*(m-32t)) (<=1 in masked region), colf = exp(s*(n&31)) <= e^22.
  const float colf0 = __expf(s * (float)cc);         // jj = 0
  const float colf1 = __expf(s * (float)(16 + cc));  // jj = 1
  bf16_t* sw = swA + (size_t)wid * 2 * 32 * 16;
  for (int t = 0; t <= wid; ++t) {
    v8f sacc[2][2];
#pragma unroll
    for (int i = 0; i < 2; ++i)
#pragma unroll
      for (int jj = 0; jj < 2; ++jj) sacc[i][jj] = vzero();
#pragma unroll
    for (int kk = 0; kk < 4; ++kk) {
      const v16bf a0 = frag_ld(qA, (2 * wid) * 4 + kk);
      const v16bf a1 = frag_ld(qA, (2 * wid + 1) * 4 + kk);
#pragma unroll
      for (int jj = 0; jj < 2; ++jj) {
        const v16bf bT = frag_ld(kB, kk * 16 + 2 * t + jj);
        sacc[0][jj] = wmma_bf16(a0, bT, sacc[0][jj]);
        sacc[1][jj] = wmma_bf16(a1, bT, sacc[1][jj]);
      }
    }
    // causal decay mask; scatter C-layout -> A-frag layout in per-wave LDS
#pragma unroll
    for (int i = 0; i < 2; ++i)
#pragma unroll
      for (int r = 0; r < 8; ++r) {
        const int m = m0 + i * 16 + cr + r;
        const float rowf = __expf(-s * (float)(m - 32 * t));
        const int dl = (cr + r) + 16 * (cc >> 3);
#pragma unroll
        for (int jj = 0; jj < 2; ++jj) {
          const int n = t * 32 + jj * 16 + cc;
          const float colf = jj ? colf1 : colf0;
          const float v = (m >= n) ? sacc[i][jj][r] * rowf * colf : 0.0f;
          sw[(size_t)(i * 32 + dl) * 16 + 8 * jj + (cc & 7)] = (bf16_t)v;
        }
      }
    // O += S(32x32) @ V(32x128)  (wave-private; compiler inserts s_wait_dscnt)
#pragma unroll
    for (int i = 0; i < 2; ++i) {
      const v16bf aS = frag_ld(sw, i);
#pragma unroll
      for (int j = 0; j < 8; ++j) {
        const v16bf bv = frag_ld(vB, t * 8 + j);
        o[i][j] = wmma_bf16(aS, bv, o[i][j]);
      }
    }
  }

  // store: attn[token, h*HD + e]
  float* ob = attn + (size_t)(c * CHUNK) * (NH * HD) + h * HD;
#pragma unroll
  for (int i = 0; i < 2; ++i)
#pragma unroll
    for (int j = 0; j < 8; ++j)
#pragma unroll
      for (int r = 0; r < 8; ++r) {
        const int m = m0 + i * 16 + cr + r;
        const int e = j * 16 + cc;
        ob[(size_t)m * (NH * HD) + e] = o[i][j][r];
      }
}

// ---------------- RMSNorm * g_w * sigmoid(gate), in place ----------------
__global__ __launch_bounds__(256) void norm_gate_kernel(
    float* __restrict__ attn, const float* __restrict__ gate,
    const float* __restrict__ gw) {
  __shared__ float red[8];
  const int n = blockIdx.x;
  const int tid = threadIdx.x;
  float* row = attn + (size_t)n * HIDDIM;
  const float* grow = gate + (size_t)n * HIDDIM;
  float vals[8];
  float ss = 0.0f;
#pragma unroll
  for (int i = 0; i < 8; ++i) {
    const float v = row[tid + i * 256];
    vals[i] = v;
    ss += v * v;
  }
#pragma unroll
  for (int off = 16; off > 0; off >>= 1) ss += __shfl_down(ss, off, 32);
  if ((tid & 31) == 0) red[tid >> 5] = ss;
  __syncthreads();
  if (tid == 0) {
    float tot = 0.0f;
    for (int i = 0; i < 8; ++i) tot += red[i];
    red[0] = tot;
  }
  __syncthreads();
  const float rms = rsqrtf(red[0] * (1.0f / (float)HIDDIM) + 1e-5f);
#pragma unroll
  for (int i = 0; i < 8; ++i) {
    const int j = tid + i * 256;
    const float sig = 1.0f / (1.0f + __expf(-grow[j]));
    row[j] = vals[i] * rms * gw[j] * sig;
  }
}

// ---------------- host-side orchestration ----------------
extern "C" void kernel_launch(void* const* d_in, const int* in_sizes, int n_in,
                              void* d_out, int out_size, void* d_ws, size_t ws_size,
                              hipStream_t stream) {
  (void)in_sizes; (void)n_in; (void)out_size; (void)ws_size;
  const float* hidden = (const float*)d_in[0];
  const int*   pos    = (const int*)d_in[1];
  const float* Wqkv   = (const float*)d_in[2];
  const float* Wg     = (const float*)d_in[3];
  const float* Wd     = (const float*)d_in[4];
  const float* gw     = (const float*)d_in[5];
  float* out = (float*)d_out;
  float* ws  = (float*)d_ws;

  // workspace layout (floats): qkv | gate | kvc | kvin | attn(->y in place)
  float* qkv   = ws;
  float* gateb = qkv   + (size_t)NTOK * QKVW;                // 192 MiB
  float* kvc   = gateb + (size_t)NTOK * HIDDIM;              //  64 MiB
  float* kvin  = kvc   + (size_t)NCHUNK * NH * HD * HD;      //  32 MiB
  float* attnb = kvin  + (size_t)NCHUNK * NH * HD * HD;      //  32 MiB (+64)

  const dim3 thr(256);
  gemm_bf16_kernel<QKVW, HIDDIM>
      <<<dim3(QKVW / 128, NTOK / 128), thr, 0, stream>>>(hidden, Wqkv, qkv);
  gemm_bf16_kernel<HIDDIM, HIDDIM>
      <<<dim3(HIDDIM / 128, NTOK / 128), thr, 0, stream>>>(hidden, Wg, gateb);
  rope_kernel<<<(NTOK * NH * 64) / 256, thr, 0, stream>>>(qkv, pos);
  kv_chunk_kernel<<<dim3(NH, NCHUNK), thr, 0, stream>>>(qkv, kvc);
  kv_scan_kernel<<<(NH * HD * HD) / 256, thr, 0, stream>>>(kvc, kvin);
  const size_t smem =
      (size_t)(16 * 4 * 32 * 16 + 4 * 16 * 32 * 16 + 8 * 8 * 32 * 16 +
               4 * 8 * 32 * 16 + 8 * 2 * 32 * 16) * sizeof(bf16_t);  // 240KB
  attn_out_kernel<<<dim3(NH, NCHUNK), thr, smem, stream>>>(qkv, kvin, attnb);
  norm_gate_kernel<<<NTOK, thr, 0, stream>>>(attnb, gateb, gw);
  gemm_bf16_kernel<HIDDIM, HIDDIM>
      <<<dim3(HIDDIM / 128, NTOK / 128), thr, 0, stream>>>(attnb, Wd, out);
}
